// AdaptiveWeightInterpolationModule_15393162789087
// MI455X (gfx1250) — matrix operand order repaired and checked by reading, not compiled, not run on hardware
//
#include <hip/hip_runtime.h>
#include <hip/hip_bf16.h>
#include <math.h>

// ---- module hyperparameters ----
#define SIGMA_D   1.5f
#define TAO_R     0.05f
#define LAMBDA_E  (30.0f / 255.0f)
#define THRESH_   1e-6f
#define EPS_      1e-6f

// ---- problem dimensions (compile-time so /,% lower to mul-shift) ----
constexpr int H_  = 720;
constexpr int W_  = 1280;
constexpr int HW_ = H_ * W_;
constexpr int TILE_   = 16;
constexpr int TILESX_ = W_ / TILE_;   // 80
constexpr int TILESY_ = H_ / TILE_;   // 45
constexpr int WPB_    = 8;            // waves per block
constexpr int SROW_   = 19;           // padded LDS row stride for 18x18 region

typedef __attribute__((ext_vector_type(16))) _Float16 v16h;
typedef __attribute__((ext_vector_type(8)))  float    v8f;

#define ATOMIC_FADD(p, v) \
  (void)__hip_atomic_fetch_add((p), (v), __ATOMIC_RELAXED, __HIP_MEMORY_SCOPE_AGENT)

// ---------------------------------------------------------------------------
// 0) zero accumulator planes
// ---------------------------------------------------------------------------
__global__ void zero_kernel(float* __restrict__ p, size_t n) {
  size_t i = (size_t)blockIdx.x * blockDim.x + threadIdx.x;
  size_t stride = (size_t)gridDim.x * blockDim.x;
  for (; i < n; i += stride) p[i] = 0.0f;
}

// ---------------------------------------------------------------------------
// 1) backward bilinear warp + photometric error (mean |i1 - warp(i2)| over C)
// ---------------------------------------------------------------------------
__global__ void warp_err_kernel(const float* __restrict__ i1,
                                const float* __restrict__ i2,
                                const float* __restrict__ flow,
                                float* __restrict__ err, int B) {
  int N = B * HW_;
  for (int idx = blockIdx.x * blockDim.x + threadIdx.x; idx < N;
       idx += gridDim.x * blockDim.x) {
    int b = idx / HW_;
    int p = idx - b * HW_;
    int y = p / W_;
    int x = p - y * W_;
    const float* fb = flow + (size_t)b * 2 * HW_;
    float u = fb[p];
    float v = fb[HW_ + p];
    float gx = fminf(fmaxf((float)x + u, 0.0f), (float)(W_ - 1));
    float gy = fminf(fmaxf((float)y + v, 0.0f), (float)(H_ - 1));
    float x0f = floorf(gx), y0f = floorf(gy);
    int x0 = (int)x0f, y0 = (int)y0f;
    int x1 = min(x0 + 1, W_ - 1), y1 = min(y0 + 1, H_ - 1);
    float wx = gx - x0f, wy = gy - y0f;
    float w00 = (1.0f - wx) * (1.0f - wy);
    float w01 = wx * (1.0f - wy);
    float w10 = (1.0f - wx) * wy;
    float w11 = wx * wy;
    const float* ib1 = i1 + (size_t)b * 3 * HW_;
    const float* ib2 = i2 + (size_t)b * 3 * HW_;
    float e = 0.0f;
#pragma unroll
    for (int c = 0; c < 3; ++c) {
      const float* ch = ib2 + (size_t)c * HW_;
      float s = ch[y0 * W_ + x0] * w00 + ch[y0 * W_ + x1] * w01 +
                ch[y1 * W_ + x0] * w10 + ch[y1 * W_ + x1] * w11;
      e += fabsf(ib1[(size_t)c * HW_ + p] - s);
    }
    err[idx] = e * (1.0f / 3.0f);
  }
}

// ---------------------------------------------------------------------------
// 2) 3x3 box filter (SAME, zero pad) + fw = exp(-(err_box/LAMBDA_E)^2)
//    One wave per full 16x16 tile (720 and 1280 are multiples of 16).
//    Horizontal 3-sum from an LDS-staged 18x18 halo region (VALU),
//    vertical 3-sum as one WMMA with a constant banded A matrix:
//        D(16x16,f32) = A(16x32,f16) x B(32x16,f16)
// ---------------------------------------------------------------------------
__global__ void box_fw_wmma_kernel(const float* __restrict__ err,
                                   float* __restrict__ fw, int B) {
  __shared__ float s[WPB_][18 * SROW_];
  int lane  = threadIdx.x & 31;
  int wslot = threadIdx.x >> 5;

  int totalTiles = B * TILESX_ * TILESY_;
  int tile = blockIdx.x * WPB_ + wslot;
  if (tile >= totalTiles) tile = totalTiles - 1;  // benign duplicate work
  constexpr int TPB = TILESX_ * TILESY_;
  int b  = tile / TPB;
  int t  = tile - b * TPB;
  int ty = t / TILESX_;
  int tx = t - ty * TILESX_;
  int r0 = ty * TILE_ - 1;  // halo origin (row)
  int c0 = tx * TILE_ - 1;  // halo origin (col)

  const float* eb = err + (size_t)b * HW_;
  // cooperative stage of the 18x18 halo region, zero outside image
  for (int i = lane; i < 18 * 18; i += 32) {
    int rr = i / 18, cc = i - rr * 18;
    int r = r0 + rr, c = c0 + cc;
    float v = 0.0f;
    if (r >= 0 && r < H_ && c >= 0 && c < W_) v = eb[(size_t)r * W_ + c];
    s[wslot][rr * SROW_ + cc] = v;
  }
  __syncthreads();

  int n = lane & 15;
  int kbase = (lane < 16) ? 0 : 16;
  v16h amat, bmat;
#pragma unroll
  for (int e = 0; e < 16; ++e) {
    // A: lanes 0-15 -> M=lane, K = e; lanes 16-31 -> K = 16 + e
    int K = e + (lane < 16 ? 0 : 16);
    int M = lane & 15;
    amat[e] = (_Float16)((K >= M && K <= M + 2) ? 1.0f : 0.0f);
    // B: N = lane&15, K = kbase + e; rows 0..17 hold horizontal 3-sums
    int Kb = kbase + e;
    float h = 0.0f;
    if (Kb < 18) {
      const float* row = &s[wslot][Kb * SROW_];
      h = row[n] + row[n + 1] + row[n + 2];
    }
    bmat[e] = (_Float16)h;
  }

  v8f acc = {};
  acc = __builtin_amdgcn_wmma_f32_16x16x32_f16(
      /*neg_a=*/false, amat, /*neg_b=*/false, bmat,
      /*c_mod=*/(short)0, acc, /*reuse_a=*/false, /*reuse_b=*/false);

  // D layout: element j -> M = j (lanes 0-15) or j+8 (lanes 16-31), N = lane&15
  float* fb = fw + (size_t)b * HW_ +
              (size_t)(ty * TILE_ + ((lane < 16) ? 0 : 8)) * W_ +
              (tx * TILE_ + n);
  const float inv9  = 1.0f / 9.0f;
  const float invLe = 1.0f / LAMBDA_E;
#pragma unroll
  for (int j = 0; j < 8; ++j) {
    float be = acc[j] * inv9;
    float tq = be * invLe;
    fb[(size_t)j * W_] = __expf(-tq * tq);  // LAMBDA_V == 1
  }
}

// ---------------------------------------------------------------------------
// 3) forward gaussian splat: scatter img*fw (3ch), fw, 1 into 5 planes
//    taps dy,dx in {-1,0,1,2} around floor(target); truncated at TAO_R
// ---------------------------------------------------------------------------
__global__ void splat_kernel(const float* __restrict__ img,
                             const float* __restrict__ flow,
                             const float* __restrict__ fw,
                             float* __restrict__ acc,  // [5][B*H*W]
                             int B) {
  int N = B * HW_;
  const float inv2s2 = 1.0f / (2.0f * SIGMA_D * SIGMA_D);
  for (int idx = blockIdx.x * blockDim.x + threadIdx.x; idx < N;
       idx += gridDim.x * blockDim.x) {
    int b = idx / HW_;
    int p = idx - b * HW_;
    int y = p / W_;
    int x = p - y * W_;
    const float* fb = flow + (size_t)b * 2 * HW_;
    float u = fb[p], v = fb[HW_ + p];
    float txf = (float)x + u;
    float tyf = (float)y + v;
    float fx = floorf(txf), fy = floorf(tyf);
    int ifx = (int)fx, ify = (int)fy;
    float wv = fw[idx];
    const float* ib = img + (size_t)b * 3 * HW_;
    float s0 = ib[p] * wv;
    float s1 = ib[HW_ + p] * wv;
    float s2 = ib[2 * HW_ + p] * wv;
#pragma unroll
    for (int dy = -1; dy <= 2; ++dy) {
      int iy = ify + dy;
      if (iy < 0 || iy >= H_) continue;
      float dyf = tyf - (float)iy;
#pragma unroll
      for (int dx = -1; dx <= 2; ++dx) {
        int ix = ifx + dx;
        if (ix < 0 || ix >= W_) continue;
        float dxf = txf - (float)ix;
        float d2 = dxf * dxf + dyf * dyf;
        float g = __expf(-d2 * inv2s2);
        if (g > TAO_R) {
          size_t o = (size_t)b * HW_ + (size_t)iy * W_ + ix;
          ATOMIC_FADD(&acc[o], s0 * g);
          ATOMIC_FADD(&acc[(size_t)N + o], s1 * g);
          ATOMIC_FADD(&acc[(size_t)2 * N + o], s2 * g);
          ATOMIC_FADD(&acc[(size_t)3 * N + o], wv * g);
          ATOMIC_FADD(&acc[(size_t)4 * N + o], g);
        }
      }
    }
  }
}

// ---------------------------------------------------------------------------
// 4) final adaptive blend
// ---------------------------------------------------------------------------
__global__ void combine_kernel(const float* __restrict__ acc1,
                               const float* __restrict__ acc2,
                               float* __restrict__ out, int B) {
  int N = B * HW_;
  for (int idx = blockIdx.x * blockDim.x + threadIdx.x; idx < N;
       idx += gridDim.x * blockDim.x) {
    float pw1 = acc1[(size_t)3 * N + idx];
    float rw1 = acc1[(size_t)4 * N + idx];
    float pw2 = acc2[(size_t)3 * N + idx];
    float rw2 = acc2[(size_t)4 * N + idx];
    float w1 = pw1 / (rw1 + THRESH_);
    float w2 = pw2 / (rw2 + THRESH_);
    float dn = 1.0f / (w1 + w2 + EPS_);
    float m1 = w1 / (pw1 + THRESH_);  // i1*w1 = p1 * m1
    float m2 = w2 / (pw2 + THRESH_);
    int b = idx / HW_;
    int p = idx - b * HW_;
#pragma unroll
    for (int c = 0; c < 3; ++c) {
      float p1 = acc1[(size_t)c * N + idx];
      float p2 = acc2[(size_t)c * N + idx];
      out[((size_t)b * 3 + c) * HW_ + p] = (p1 * m1 + p2 * m2) * dn;
    }
  }
}

// ---------------------------------------------------------------------------
extern "C" void kernel_launch(void* const* d_in, const int* in_sizes, int n_in,
                              void* d_out, int out_size, void* d_ws, size_t ws_size,
                              hipStream_t stream) {
  const float* in1 = (const float*)d_in[0];
  const float* in2 = (const float*)d_in[1];
  const float* fl1 = (const float*)d_in[2];
  const float* fl2 = (const float*)d_in[3];
  float* out = (float*)d_out;

  int B = in_sizes[0] / (3 * HW_);
  size_t N = (size_t)B * HW_;

  // workspace layout (floats): err[N] | fw[N] | acc1[5N] | acc2[5N]
  float* ws   = (float*)d_ws;
  float* err  = ws;
  float* fwb  = ws + N;
  float* acc1 = ws + 2 * N;
  float* acc2 = ws + 7 * N;

  int totalTiles = B * TILESX_ * TILESY_;

  dim3 blk(256);
  int nBlocks   = (int)((N + 255) / 256);
  int boxBlocks = (totalTiles + WPB_ - 1) / WPB_;
  int zBlocks   = (int)((10 * N + 255) / 256);

  zero_kernel<<<zBlocks, blk, 0, stream>>>(acc1, 10 * N);

  // branch 1: warp frame1 forward with flow input3
  warp_err_kernel<<<nBlocks, blk, 0, stream>>>(in1, in2, fl1, err, B);
  box_fw_wmma_kernel<<<boxBlocks, blk, 0, stream>>>(err, fwb, B);
  splat_kernel<<<nBlocks, blk, 0, stream>>>(in1, fl1, fwb, acc1, B);

  // branch 2: warp frame2 forward with flow input4
  warp_err_kernel<<<nBlocks, blk, 0, stream>>>(in2, in1, fl2, err, B);
  box_fw_wmma_kernel<<<boxBlocks, blk, 0, stream>>>(err, fwb, B);
  splat_kernel<<<nBlocks, blk, 0, stream>>>(in2, fl2, fwb, acc2, B);

  combine_kernel<<<nBlocks, blk, 0, stream>>>(acc1, acc2, out, B);
}
